// AttentionBlock_24739011625024
// MI455X (gfx1250) — matrix operand order repaired and checked by reading, not compiled, not run on hardware
//
#include <hip/hip_runtime.h>
#include <hip/hip_bf16.h>
#include <math.h>

// ---------------------------------------------------------------------------
// AttentionBlock for MI455X (gfx1250): f16 WMMA + async-to-LDS + TDM staging.
//   1) convert x, Wq, Wk, Wv -> f16 in d_ws
//   2) WMMA GEMM: W tile staged via GLOBAL_LOAD_ASYNC_TO_LDS_B128 (double
//      buffered, s_wait_asynccnt pipelining). Q, K row-major, V transposed.
//   3) flash attention: K tiles staged by the Tensor Data Mover
//      (tensor_load_to_lds descriptor DMA, s_wait_tensorcnt, LDS row padding
//      for bank-conflict-free B-fragment reads), online softmax, all matmuls
//      via v_wmma_f32_16x16x32_f16.
// Workspace layout (halfs): xh | wqh | wkh | wvh | qh | kh | vTh  (~68.7 MB)
// ---------------------------------------------------------------------------

typedef _Float16 half_t;
typedef __attribute__((ext_vector_type(16))) _Float16 v16h;
typedef __attribute__((ext_vector_type(8)))  _Float16 v8h;
typedef __attribute__((ext_vector_type(8)))  float    v8f;
typedef __attribute__((ext_vector_type(4)))  unsigned int u32x4;
typedef __attribute__((ext_vector_type(8)))  int          i32x8;
typedef __attribute__((ext_vector_type(4)))  int          i32x4;

#define D_MODEL 512
#define SEQ     2048
#define BATCH   8
#define BS_ROWS (BATCH * SEQ)          // 16384
#define KPITCH  520                    // 512 halfs + 16B TDM pad per K row

static __device__ __forceinline__ v16h join8(v8h lo, v8h hi) {
  v16h r;
#pragma unroll
  for (int i = 0; i < 8; ++i) { r[i] = lo[i]; r[i + 8] = hi[i]; }
  return r;
}

// --- gfx1250 async copy: 16 bytes global -> LDS, tracked by ASYNCcnt -------
static __device__ __forceinline__ void async_copy_b128(const half_t* g, half_t* l) {
  const unsigned long long ga = (unsigned long long)(uintptr_t)g;
  const unsigned int       la = (unsigned int)(uintptr_t)l;  // addr[31:0] == LDS offset
  asm volatile("global_load_async_to_lds_b128 %0, %1, off"
               :: "v"(la), "v"(ga) : "memory");
}
static __device__ __forceinline__ void wait_async_le2() {
  asm volatile("s_wait_asynccnt 0x2" ::: "memory");
}
static __device__ __forceinline__ void wait_async_le0() {
  asm volatile("s_wait_asynccnt 0x0" ::: "memory");
}

// --- gfx1250 TDM: 2D tile (tile_w elems x tile_h rows, 2B elems) -> LDS ----
// LDS rows padded by 16B every 1024B (pad_interval=7, pad_amount=3) so the
// destination pitch is tile_w + 8 halfs when tile_w == 512.
static __device__ __forceinline__ void tdm_load_2d_f16(const half_t* gsrc,
                                                       half_t* ldst,
                                                       unsigned tile_w,
                                                       unsigned tile_h,
                                                       unsigned stride_elems,
                                                       int pad_rows) {
  const unsigned long long ga = (unsigned long long)(uintptr_t)gsrc;
  const unsigned int       la = (unsigned int)(uintptr_t)ldst;
  u32x4 g0;
  g0[0] = 1u;                                          // count=1 valid D#
  g0[1] = la;                                          // lds_addr (bytes)
  g0[2] = (unsigned int)ga;                            // global_addr[31:0]
  g0[3] = (unsigned int)((ga >> 32) & 0x01FFFFFFu) | (2u << 30); // addr[56:32]|type=2
  i32x8 g1;
  g1[0] = (int)((1u << 16) |                           // data_size = 2 bytes
                (pad_rows ? ((1u << 20) | (7u << 22) | (3u << 25)) : 0u));
  g1[1] = (int)((tile_w & 0xFFFFu) << 16);             // tensor_dim0[15:0]
  g1[2] = (int)(((tile_w >> 16) & 0xFFFFu) |           // tensor_dim0[31:16]
                ((tile_h & 0xFFFFu) << 16));           // tensor_dim1[15:0]
  g1[3] = (int)(((tile_h >> 16) & 0xFFFFu) |           // tensor_dim1[31:16]
                ((tile_w & 0xFFFFu) << 16));           // tile_dim0
  g1[4] = (int)(tile_h & 0xFFFFu);                     // tile_dim1 (tile_dim2=0)
  g1[5] = (int)stride_elems;                           // tensor_dim0_stride[31:0]
  g1[6] = 0;
  g1[7] = 0;
  const i32x4 z4 = {0, 0, 0, 0};
  const i32x8 z8 = {0, 0, 0, 0, 0, 0, 0, 0};
  __builtin_amdgcn_tensor_load_to_lds(g0, g1, z4, z4, z8, 0);
}
static __device__ __forceinline__ void wait_tensor_le1() {
  __builtin_amdgcn_s_wait_tensorcnt(1);
}
static __device__ __forceinline__ void wait_tensor_le0() {
  __builtin_amdgcn_s_wait_tensorcnt(0);
}

// -------------------------------- convert ---------------------------------
__global__ void cvt_f32_to_f16_kernel(const float* __restrict__ in,
                                      half_t* __restrict__ out, int n) {
  int i = blockIdx.x * blockDim.x + threadIdx.x;
  if (i < n) out[i] = (half_t)in[i];
}

// ------------------------------- QKV GEMM ----------------------------------
// C[m,n] = sum_d X[m,d] * W[n,d].  Block = 4 waves, tile 128(M) x 64(N).
// W tile (64 x 32 f16) staged in LDS via async loads, double buffered.
__global__ __launch_bounds__(128, 1)
void qkv_gemm_kernel(const half_t* __restrict__ X, const half_t* __restrict__ W,
                     half_t* __restrict__ out, int transpose_v) {
  __shared__ __align__(16) half_t lds_w[2][64][32];   // 8 KB, double buffered

  const int tid     = threadIdx.x;
  const int lane    = tid & 31;
  const int wave    = tid >> 5;
  const int half_id = lane >> 4;
  const int l16     = lane & 15;

  const int m0 = blockIdx.x * 128 + wave * 32;
  const int n0 = blockIdx.y * 64;

  const int klo  = half_id * 8;
  const int kb16 = half_id * 16;
  const half_t* xr0 = X + (size_t)(m0 + l16) * D_MODEL;
  const half_t* xr1 = X + (size_t)(m0 + 16 + l16) * D_MODEL;

  const int c0row = tid >> 2,          c0off = (tid & 3) * 8;
  const int c1row = (tid + 128) >> 2,  c1off = ((tid + 128) & 3) * 8;

#define STAGE_W(buf, d0)                                                        \
  do {                                                                          \
    async_copy_b128(W + (size_t)(n0 + c0row) * D_MODEL + (d0) + c0off,          \
                    &lds_w[buf][c0row][c0off]);                                 \
    async_copy_b128(W + (size_t)(n0 + c1row) * D_MODEL + (d0) + c1off,          \
                    &lds_w[buf][c1row][c1off]);                                 \
  } while (0)

#define LOAD_A(dst, d0)                                                         \
  do {                                                                          \
    dst[0] = join8(*(const v8h*)(xr0 + (d0) + klo),                             \
                   *(const v8h*)(xr0 + (d0) + klo + 16));                       \
    dst[1] = join8(*(const v8h*)(xr1 + (d0) + klo),                             \
                   *(const v8h*)(xr1 + (d0) + klo + 16));                       \
  } while (0)

  v8f acc[2][4] = {};
  v16h aCur[2], aNxt[2];

  STAGE_W(0, 0);
  LOAD_A(aCur, 0);

#pragma unroll 4
  for (int it = 0; it < D_MODEL / 32; ++it) {
    const int d0  = it * 32;
    const int buf = it & 1;

    __syncthreads();
    if (it + 1 < D_MODEL / 32) {
      STAGE_W(buf ^ 1, d0 + 32);
      wait_async_le2();
    } else {
      wait_async_le0();
    }
    __syncthreads();

    if (it + 1 < D_MODEL / 32) LOAD_A(aNxt, d0 + 32);

#pragma unroll
    for (int t = 0; t < 4; ++t) {
      const int n = t * 16 + l16;
      const v16h b = join8(*(const v8h*)(&lds_w[buf][n][klo]),
                           *(const v8h*)(&lds_w[buf][n][klo + 16]));
      acc[0][t] = __builtin_amdgcn_wmma_f32_16x16x32_f16(
          false, aCur[0], false, b, (short)0, acc[0][t], false, false);
      acc[1][t] = __builtin_amdgcn_wmma_f32_16x16x32_f16(
          false, aCur[1], false, b, (short)0, acc[1][t], false, false);
    }
    aCur[0] = aNxt[0];
    aCur[1] = aNxt[1];
  }

#pragma unroll
  for (int rt = 0; rt < 2; ++rt)
#pragma unroll
    for (int t = 0; t < 4; ++t) {
      const int n = n0 + t * 16 + l16;
#pragma unroll
      for (int v = 0; v < 8; ++v) {
        const int row = m0 + rt * 16 + v + half_id * 8;
        const half_t val = (half_t)acc[rt][t][v];
        if (!transpose_v) {
          out[(size_t)row * D_MODEL + n] = val;
        } else {
          const int b = row >> 11;
          const int s = row & (SEQ - 1);
          out[((size_t)b * D_MODEL + n) * SEQ + s] = val;
        }
      }
    }
#undef STAGE_W
#undef LOAD_A
}

// ---------------------------- flash attention ------------------------------
// One block = 16 query rows of one batch.  TDM double-buffers the 32-key K
// tile into padded LDS; 4 waves split the d=512 score reduction (partials
// summed via LDS) and split e=512 for the PV product.
__global__ __launch_bounds__(128, 1)
void flash_attn_kernel(const half_t* __restrict__ Q, const half_t* __restrict__ K,
                       const half_t* __restrict__ VT, float* __restrict__ O,
                       float scale) {
  __shared__ __align__(16) half_t lds_k[2][32][KPITCH];  // ~65 KB, TDM dest
  __shared__ __align__(16) float  lds_s[4][16][32];
  __shared__ __align__(16) half_t lds_p[4][16][32];

  const int lane    = threadIdx.x & 31;
  const int wave    = threadIdx.x >> 5;
  const int half_id = lane >> 4;
  const int l16     = lane & 15;

  const int tile = blockIdx.x;                 // 0 .. 1023
  const int b    = tile >> 7;
  const int m0   = (tile & 127) * 16;

  const half_t* Qb = Q  + (size_t)b * SEQ * D_MODEL;
  const half_t* Kb = K  + (size_t)b * SEQ * D_MODEL;
  const half_t* Vb = VT + (size_t)b * D_MODEL * SEQ;

  const int arow = m0 + l16;
  const int klo  = half_id * 8;
  const int kb16 = half_id * 16;

  v16h qf[4];
#pragma unroll
  for (int c = 0; c < 4; ++c) {
    const int d0 = wave * 128 + c * 32;
    qf[c] = join8(*(const v8h*)(Qb + (size_t)arow * D_MODEL + d0 + klo),
                  *(const v8h*)(Qb + (size_t)arow * D_MODEL + d0 + klo + 16));
  }

  float mrow[8], lrow[8];
#pragma unroll
  for (int v = 0; v < 8; ++v) { mrow[v] = -3.0e38f; lrow[v] = 0.0f; }
  v8f o[8] = {};

  // prologue: DMA first K tile (32 keys x 512 d) into buffer 0
  if (wave == 0)
    tdm_load_2d_f16(Kb, &lds_k[0][0][0], D_MODEL, 32, D_MODEL, 1);

  for (int j0 = 0; j0 < SEQ; j0 += 32) {
    const int buf = (j0 >> 5) & 1;

    __syncthreads();                           // lds_k[buf^1] fully consumed
    if (wave == 0) {
      if (j0 + 32 < SEQ) {
        tdm_load_2d_f16(Kb + (size_t)(j0 + 32) * D_MODEL, &lds_k[buf ^ 1][0][0],
                        D_MODEL, 32, D_MODEL, 1);
        wait_tensor_le1();                     // older DMA (this buf) complete
      } else {
        wait_tensor_le0();
      }
    }
    __syncthreads();                           // K tile visible to all waves

    if (j0 + 32 < SEQ)                         // warm next V tile toward L2
      __builtin_prefetch(Vb + (size_t)(wave * 128 + l16) * SEQ + j0 + 32, 0, 0);

    // ---- partial scores over this wave's 128-wide d slice (K from LDS) ----
    v8f s[2] = {};
#pragma unroll
    for (int c = 0; c < 4; ++c) {
      const int d0 = wave * 128 + c * 32;
#pragma unroll
      for (int t = 0; t < 2; ++t) {
        const int key = t * 16 + l16;          // key row inside LDS tile
        const v16h bm = join8(*(const v8h*)(&lds_k[buf][key][d0 + klo]),
                              *(const v8h*)(&lds_k[buf][key][d0 + klo + 16]));
        s[t] = __builtin_amdgcn_wmma_f32_16x16x32_f16(
            false, qf[c], false, bm, (short)0, s[t], false, false);
      }
    }
#pragma unroll
    for (int t = 0; t < 2; ++t)
#pragma unroll
      for (int v = 0; v < 8; ++v)
        lds_s[wave][v + half_id * 8][l16 + t * 16] = s[t][v];
    __syncthreads();

    float sv[2][8];
#pragma unroll
    for (int t = 0; t < 2; ++t)
#pragma unroll
      for (int v = 0; v < 8; ++v) {
        float a = 0.0f;
#pragma unroll
        for (int w = 0; w < 4; ++w) a += lds_s[w][v + half_id * 8][l16 + t * 16];
        sv[t][v] = a * scale;
      }
    __syncthreads();

    float alpha[8], pr0[8], pr1[8];
#pragma unroll
    for (int v = 0; v < 8; ++v) {
      float mx = fmaxf(sv[0][v], sv[1][v]);
#pragma unroll
      for (int off = 1; off < 16; off <<= 1) mx = fmaxf(mx, __shfl_xor(mx, off, 32));
      const float mnew = fmaxf(mrow[v], mx);
      alpha[v] = __expf(mrow[v] - mnew);
      pr0[v]   = __expf(sv[0][v] - mnew);
      pr1[v]   = __expf(sv[1][v] - mnew);
      float rs = pr0[v] + pr1[v];
#pragma unroll
      for (int off = 1; off < 16; off <<= 1) rs += __shfl_xor(rs, off, 32);
      lrow[v] = lrow[v] * alpha[v] + rs;
      mrow[v] = mnew;
    }

#pragma unroll
    for (int t = 0; t < 8; ++t)
#pragma unroll
      for (int v = 0; v < 8; ++v) o[t][v] *= alpha[v];

#pragma unroll
    for (int v = 0; v < 8; ++v) {
      lds_p[wave][v + half_id * 8][l16]      = (half_t)pr0[v];
      lds_p[wave][v + half_id * 8][l16 + 16] = (half_t)pr1[v];
    }
    const v16h pA = join8(*(const v8h*)(&lds_p[wave][l16][klo]),
                          *(const v8h*)(&lds_p[wave][l16][klo + 16]));

#pragma unroll
    for (int t = 0; t < 8; ++t) {
      const int e = wave * 128 + t * 16 + l16;
      const v16h bm = join8(*(const v8h*)(Vb + (size_t)e * SEQ + j0 + klo),
                            *(const v8h*)(Vb + (size_t)e * SEQ + j0 + klo + 16));
      o[t] = __builtin_amdgcn_wmma_f32_16x16x32_f16(
          false, pA, false, bm, (short)0, o[t], false, false);
    }
  }

  float inv[8];
#pragma unroll
  for (int v = 0; v < 8; ++v) inv[v] = 1.0f / lrow[v];
#pragma unroll
  for (int t = 0; t < 8; ++t) {
    const int e = wave * 128 + t * 16 + l16;
#pragma unroll
    for (int v = 0; v < 8; ++v) {
      const int row = m0 + v + half_id * 8;
      O[((size_t)b * SEQ + row) * D_MODEL + e] = o[t][v] * inv[v];
    }
  }
}

// ------------------------------- launcher ----------------------------------
extern "C" void kernel_launch(void* const* d_in, const int* in_sizes, int n_in,
                              void* d_out, int out_size, void* d_ws, size_t ws_size,
                              hipStream_t stream) {
  (void)in_sizes; (void)n_in; (void)out_size; (void)ws_size;

  const float* x  = (const float*)d_in[0];
  const float* Wq = (const float*)d_in[1];
  const float* Wk = (const float*)d_in[2];
  const float* Wv = (const float*)d_in[3];
  float* out = (float*)d_out;

  const size_t NX = (size_t)BS_ROWS * D_MODEL;
  const size_t NW = (size_t)D_MODEL * D_MODEL;

  half_t* xh  = (half_t*)d_ws;
  half_t* wqh = xh  + NX;
  half_t* wkh = wqh + NW;
  half_t* wvh = wkh + NW;
  half_t* qh  = wvh + NW;
  half_t* kh  = qh  + NX;
  half_t* vTh = kh  + NX;

  cvt_f32_to_f16_kernel<<<(int)((NX + 255) / 256), 256, 0, stream>>>(x,  xh,  (int)NX);
  cvt_f32_to_f16_kernel<<<(int)((NW + 255) / 256), 256, 0, stream>>>(Wq, wqh, (int)NW);
  cvt_f32_to_f16_kernel<<<(int)((NW + 255) / 256), 256, 0, stream>>>(Wk, wkh, (int)NW);
  cvt_f32_to_f16_kernel<<<(int)((NW + 255) / 256), 256, 0, stream>>>(Wv, wvh, (int)NW);

  dim3 ggemm(BS_ROWS / 128, D_MODEL / 64);       // (128, 8)
  qkv_gemm_kernel<<<ggemm, 128, 0, stream>>>(xh, wqh, qh,  0);
  qkv_gemm_kernel<<<ggemm, 128, 0, stream>>>(xh, wkh, kh,  0);
  qkv_gemm_kernel<<<ggemm, 128, 0, stream>>>(xh, wvh, vTh, 1);

  const float scale = 0.044194173824159216f;     // 1/sqrt(512)
  flash_attn_kernel<<<BATCH * (SEQ / 16), 128, 0, stream>>>(qh, kh, vTh, out, scale);
}
